// LSTMPolicy_21474836480421
// MI455X (gfx1250) — compile-verified, matrix operand order checked by reading, not run on hardware
//
#include <hip/hip_runtime.h>
#include <cstdint>
#include <cstddef>

#define B_    64
#define T_    2048
#define OBS_  64
#define ACT_  8
#define H_    256
#define G4_   1024
#define HALF_LOG_2PI 0.9189385332046727f

typedef __attribute__((ext_vector_type(16))) __bf16 v16bf;
typedef __attribute__((ext_vector_type(8)))  float  v8f;

// ---------- bf16 helpers (bit-exact storage in unsigned short) ----------
__device__ __forceinline__ unsigned short f2b(float f) {
  union { float f; unsigned int u; } v; v.f = f;
  unsigned int r = v.u + 0x7FFFu + ((v.u >> 16) & 1u);   // RNE
  return (unsigned short)(r >> 16);
}
__device__ __forceinline__ float b2f(unsigned short b) {
  union { unsigned int u; float f; } v; v.u = ((unsigned int)b) << 16;
  return v.f;
}

// ---------- branch-free hardware transcendentals (gfx1250 v_tanh_f32) ----
__device__ __forceinline__ float fast_tanh(float x) {
#if __has_builtin(__builtin_amdgcn_tanhf)
  return __builtin_amdgcn_tanhf(x);
#else
  float y;
  asm volatile("v_tanh_f32 %0, %1\n\tv_nop\n\tv_nop" : "=v"(y) : "v"(x));
  return y;
#endif
}
__device__ __forceinline__ float fast_sigmoid(float x) {
  return __builtin_fmaf(0.5f, fast_tanh(0.5f * x), 0.5f);
}

// ---------- async global->LDS copy (CDNA5, tracked by ASYNCcnt) ----------
// LDS byte address comes from the VDST VGPR; GV mode: 64-bit vaddr, saddr = off
__device__ __forceinline__ void async_load_b128_to_lds(unsigned lds_byte_off,
                                                       const void* gaddr) {
  asm volatile("global_load_async_to_lds_b128 %0, %1, off"
               :: "v"(lds_byte_off), "v"((unsigned long long)(size_t)gaddr)
               : "memory");
}
__device__ __forceinline__ void wait_asynccnt0() {
  asm volatile("s_wait_asynccnt 0x0" ::: "memory");
}

// 16-bit operand fragment K mapping (per CDNA5 ISA 7.12.2, 16-bit A 16x32):
// lane in [0,16): K base 0 ; lane in [16,32): K base +8 (then +16 for VGPRs 4..7)
__device__ __forceinline__ int klocal_of(int lane, int e) {
  int hiL = lane >> 4;
  int v = e >> 1, pos = e & 1;
  int base = (v < 4) ? (2 * v) : (16 + 2 * (v - 4));
  return base + pos + 8 * hiL;
}
// inverse: given k_local (0..31) -> which half-lane group + element index
__device__ __forceinline__ void inv_kl(int kl, int& hi2, int& e2) {
  hi2 = (kl >> 3) & 1;
  int kk = kl - 8 * hi2;
  int pos = kk & 1;
  int v = (kk < 8) ? (kk >> 1) : (4 + ((kk - 16) >> 1));
  e2 = 2 * v + pos;
}

// =====================================================================
// Kernel 0: convert weights fp32 -> bf16 in WMMA B-fragment order.
// Fragment layout: [frag][lane(32)][e(16)] bf16 ; frag = ntile*KSTEPS + kstep
// B[k][n] = W[n][k]  (gates = h @ W^T)
// =====================================================================
__global__ __launch_bounds__(256) void prep_weights_kernel(
    const float* __restrict__ Wp, const float* __restrict__ W_ih,
    const float* __restrict__ W_hh, const float* __restrict__ Wa,
    const float* __restrict__ Wc,
    unsigned short* __restrict__ wpF, unsigned short* __restrict__ wihF,
    unsigned short* __restrict__ whhF, unsigned short* __restrict__ wawcF) {
  const int NP = 16 * 2 * 512;     // Wp : 16 ntiles (H), 2 ksteps (OBS)
  const int NI = 64 * 8 * 512;     // W_ih: 64 ntiles (4H), 8 ksteps (H)
  const int NH = 64 * 8 * 512;     // W_hh: same shape
  const int NW = 8 * 512;          // [Wa;Wc;0pad]: 1 ntile, 8 ksteps
  const int total = NP + NI + NH + NW;
  for (int idx = blockIdx.x * blockDim.x + threadIdx.x; idx < total;
       idx += gridDim.x * blockDim.x) {
    int loc, sect;
    if (idx < NP)                { loc = idx;                sect = 0; }
    else if (idx < NP + NI)      { loc = idx - NP;           sect = 1; }
    else if (idx < NP + NI + NH) { loc = idx - NP - NI;      sect = 2; }
    else                         { loc = idx - NP - NI - NH; sect = 3; }
    int frag = loc >> 9, rem = loc & 511;
    int lane = rem >> 4, e = rem & 15;
    int nl = lane & 15;
    int kl = klocal_of(lane, e);
    float srcv;
    if (sect == 0) {
      int n = frag >> 1, ks = frag & 1;
      srcv = Wp[(size_t)(n * 16 + nl) * OBS_ + ks * 32 + kl];
      wpF[loc] = f2b(srcv);
    } else if (sect == 1) {
      int n = frag >> 3, ks = frag & 7;
      srcv = W_ih[(size_t)(n * 16 + nl) * H_ + ks * 32 + kl];
      wihF[loc] = f2b(srcv);
    } else if (sect == 2) {
      int n = frag >> 3, ks = frag & 7;
      srcv = W_hh[(size_t)(n * 16 + nl) * H_ + ks * 32 + kl];
      whhF[loc] = f2b(srcv);
    } else {
      int ks = frag;           // 0..7
      int k = ks * 32 + kl;
      srcv = (nl < ACT_) ? Wa[(size_t)nl * H_ + k]
                         : ((nl == ACT_) ? Wc[k] : 0.0f);
      wawcF[loc] = f2b(srcv);
    }
  }
}

// =====================================================================
// Kernel 1: fused input projection + pre-gate GEMM.
//   x   = tanh(obs @ Wp^T + bp)                 (WMMA bf16, K=64)
//   pre = x @ W_ih^T + (b_ih + b_hh)            (WMMA bf16, K=256)
// pre stored bf16 in [T][B][1024] layout (recurrence reads a contiguous
// 32 KB slab per step).  One block = 16 consecutive bt rows (same b).
// =====================================================================
__global__ __launch_bounds__(256) void encode_kernel(
    const float* __restrict__ obs, const float* __restrict__ bp,
    const float* __restrict__ b_ih, const float* __restrict__ b_hh,
    const unsigned short* __restrict__ wpF,
    const unsigned short* __restrict__ wihF,
    unsigned short* __restrict__ pre) {
  __shared__ unsigned short aLds[2 * 32 * 16];    // obs tile, A-frag order (K=64)
  __shared__ unsigned short xLds[8 * 32 * 16];    // x tile,  A-frag order (K=256)
  __shared__ unsigned short outS[16 * 1024];      // pre tile staging
  const v8f vz = {0.f, 0.f, 0.f, 0.f, 0.f, 0.f, 0.f, 0.f};

  int tid = threadIdx.x, lane = tid & 31, w = tid >> 5;
  int lo = lane & 15, hiL = lane >> 4;
  int m0 = blockIdx.x * 16;            // bt tile base (never crosses b: T%16==0)
  int b = m0 / T_, t0 = m0 % T_;

  // phase 0: obs tile -> bf16 A-fragments in LDS
  for (int idx = tid; idx < 2 * 32 * 16; idx += 256) {
    int ks = idx >> 9, rem = idx & 511, ln = rem >> 4, e = rem & 15;
    int row = ln & 15;
    int k = ks * 32 + klocal_of(ln, e);
    aLds[idx] = f2b(obs[(size_t)(m0 + row) * OBS_ + k]);
  }
  __syncthreads();

  // phase 1: x = tanh(obs @ Wp^T + bp); each wave does 2 of 16 N-tiles
  {
    v8f c0 = vz, c1 = vz;
#pragma unroll
    for (int ks = 0; ks < 2; ++ks) {
      v16bf a = *(const v16bf*)&aLds[(ks * 32 + lane) * 16];
      v16bf bb0 = *(const v16bf*)&wpF[(((2 * w + 0) * 2 + ks) * 32 + lane) * 16];
      v16bf bb1 = *(const v16bf*)&wpF[(((2 * w + 1) * 2 + ks) * 32 + lane) * 16];
      c0 = __builtin_amdgcn_wmma_f32_16x16x32_bf16(false, a, false, bb0, (short)0, c0, false, false);
      c1 = __builtin_amdgcn_wmma_f32_16x16x32_bf16(false, a, false, bb1, (short)0, c1, false, false);
    }
#pragma unroll
    for (int j = 0; j < 2; ++j) {
      int n = 2 * w + j;
      int col = n * 16 + lo;
      float bias = bp[col];
      int kstep2 = col >> 5, kl = col & 31;
      int hi2, e2; inv_kl(kl, hi2, e2);
#pragma unroll
      for (int r = 0; r < 8; ++r) {
        int row = r + 8 * hiL;
        float val = fast_tanh(((j == 0) ? c0[r] : c1[r]) + bias);
        xLds[(kstep2 * 32 + (row + 16 * hi2)) * 16 + e2] = f2b(val);
      }
    }
  }
  __syncthreads();

  // phase 2: pre = x @ W_ih^T + (b_ih + b_hh); each wave does 8 of 64 N-tiles
  {
    v8f cp[8];
#pragma unroll
    for (int j = 0; j < 8; ++j) cp[j] = vz;
#pragma unroll
    for (int ks = 0; ks < 8; ++ks) {
      v16bf a = *(const v16bf*)&xLds[(ks * 32 + lane) * 16];
#pragma unroll
      for (int j = 0; j < 8; ++j) {
        int n = w * 8 + j;
        v16bf bb = *(const v16bf*)&wihF[((size_t)(n * 8 + ks) * 32 + lane) * 16];
        cp[j] = __builtin_amdgcn_wmma_f32_16x16x32_bf16(false, a, false, bb, (short)0, cp[j], false, false);
      }
    }
#pragma unroll
    for (int j = 0; j < 8; ++j) {
      int col = (w * 8 + j) * 16 + lo;
      float bias = b_ih[col] + b_hh[col];
#pragma unroll
      for (int r = 0; r < 8; ++r) {
        int row = r + 8 * hiL;
        outS[row * 1024 + col] = f2b(cp[j][r] + bias);
      }
    }
  }
  __syncthreads();

  // phase 3: coalesced copy outS -> pre[t0+tl][b][:]
  {
    int tl = tid >> 4, part = tid & 15;
    const uint4* srcp = (const uint4*)&outS[tl * 1024 + part * 64];
    uint4* dstp = (uint4*)&pre[(((size_t)(t0 + tl)) * B_ + b) * G4_ + part * 64];
#pragma unroll
    for (int i = 0; i < 8; ++i) dstp[i] = srcp[i];
  }
}

// =====================================================================
// Head emission: Ch holds [mean(8 cols) | value(col 8)] for 16 rows.
// Reduce log-prob terms over 8 lanes with shfl_xor, write outputs.
// =====================================================================
__device__ __forceinline__ void emit_heads(
    const v8f& Ch, int tt, int bbase, int lo, int hiL,
    const float* __restrict__ actions, float myls, float myistd,
    float bav, float bcv, float entC, float* __restrict__ out) {
  float term[8];
#pragma unroll
  for (int r = 0; r < 8; ++r) {
    term[r] = 0.0f;
    if (lo < ACT_) {
      int b = bbase + r + 8 * hiL;
      float mean = Ch[r] + bav;
      float act = actions[((size_t)b * T_ + tt) * ACT_ + lo];
      float z = (act - mean) * myistd;
      term[r] = -0.5f * z * z - myls - HALF_LOG_2PI;
    }
  }
#pragma unroll
  for (int r = 0; r < 8; ++r) {
    term[r] += __shfl_xor(term[r], 1);
    term[r] += __shfl_xor(term[r], 2);
    term[r] += __shfl_xor(term[r], 4);
  }
  if (lo == 0) {
#pragma unroll
    for (int r = 0; r < 8; ++r) {
      int b = bbase + r + 8 * hiL;
      out[(size_t)b * T_ + tt] = term[r];                           // log_prob
      out[(size_t)B_ * T_ + (size_t)b * T_ + tt] = entC;            // entropy
    }
  }
  if (lo == ACT_) {
#pragma unroll
    for (int r = 0; r < 8; ++r) {
      int b = bbase + r + 8 * hiL;
      out[2 * (size_t)B_ * T_ + (size_t)b * T_ + tt] = Ch[r] + bcv; // value
    }
  }
}

// =====================================================================
// Kernel 2: persistent masked-LSTM recurrence + heads.
// 4 blocks x 256 threads; block handles 16 batch rows for all 2048 steps.
// h kept UNMASKED in LDS (A-fragment order); mask folded in post-GEMM:
//   gates = pre + m_row * (h_{t-1} @ W_hh^T),  c = f*(m*c) + i*g.
// Wave w owns gate tiles {q*16 + 2w + s} => i,f,g,o for one 32-wide h
// slice live in that wave's accumulators (in-register elementwise).
// pre slab staged into LDS with GLOBAL_LOAD_ASYNC_TO_LDS_B128 issued
// BEFORE the WMMA phase (latency hidden), consumed after s_wait_asynccnt.
// Each wave copies exactly the bytes it reads -> no extra barrier.
// Heads = one extra WMMA N-tile ([Wa;Wc] padded), pipelined one step back.
// =====================================================================
__global__ __launch_bounds__(256) void recurrence_kernel(
    const unsigned short* __restrict__ pre,
    const unsigned short* __restrict__ whhF,
    const unsigned short* __restrict__ wawcF,
    const int* __restrict__ done, const float* __restrict__ actions,
    const float* __restrict__ log_std, const float* __restrict__ ba,
    const float* __restrict__ bc, float* __restrict__ out) {
  __shared__ unsigned short hlds[8 * 32 * 16];     // h_{t} A-fragments (unmasked)
  __shared__ unsigned short preLds[8 * 8 * 16 * 16]; // [w][q*2+s][row][16] bf16
  __shared__ float maskLds[16];
  const v8f vz = {0.f, 0.f, 0.f, 0.f, 0.f, 0.f, 0.f, 0.f};

  int tid = threadIdx.x, lane = tid & 31, w = tid >> 5;
  int lo = lane & 15, hiL = lane >> 4;
  int bbase = blockIdx.x * 16;

  for (int idx = tid; idx < 8 * 32 * 16; idx += 256) hlds[idx] = 0;

  unsigned preBaseByte = (unsigned)(size_t)&preLds[0];  // LDS byte offset
  float myls = (lo < ACT_) ? log_std[lo] : 0.0f;
  float myistd = __expf(-myls);
  float entC = 0.0f;
#pragma unroll
  for (int a = 0; a < ACT_; ++a) entC += log_std[a];
  entC += (float)ACT_ * (0.5f + HALF_LOG_2PI);
  float bav = (lo < ACT_) ? ba[lo] : 0.0f;
  float bcv = bc[0];

  // precompute per-lane h-store LDS offsets (hidx slice of this wave)
  int hoff[2][8];
#pragma unroll
  for (int s = 0; s < 2; ++s) {
    int kl = 16 * s + lo;
    int hi2, e2; inv_kl(kl, hi2, e2);
#pragma unroll
    for (int r = 0; r < 8; ++r)
      hoff[s][r] = (w * 32 + (r + 8 * hiL + 16 * hi2)) * 16 + e2;
  }

  float cst[2][8];
#pragma unroll
  for (int s = 0; s < 2; ++s)
#pragma unroll
    for (int r = 0; r < 8; ++r) cst[s][r] = 0.0f;

  for (int t = 0; t < T_; ++t) {
    if (tid < 16)
      maskLds[tid] = 1.0f - (float)done[(size_t)(bbase + tid) * T_ + t];
    // prefetch next step's pre slab into caches (emits global_prefetch_b8)
    if (t + 1 < T_)
      __builtin_prefetch(pre + ((size_t)(t + 1) * B_ + bbase + (tid & 15)) * G4_ + (tid >> 4) * 64, 0, 1);

    // async-stage this step's pre slab into LDS (8 x b128 per wave; each
    // wave copies only the region it will itself read).
#pragma unroll
    for (int i = 0; i < 8; ++i) {
      int c = lane + 32 * i;            // 0..255
      int half = c & 1;
      int row  = (c >> 1) & 15;
      int qs   = c >> 5;                // q*2+s
      int q = qs >> 1, s = qs & 1;
      size_t gidx = ((size_t)t * B_ + bbase + row) * G4_ + q * 256 + 32 * w + 16 * s + half * 8;
      unsigned ldsOff = preBaseByte + (unsigned)(w * 2048 + qs * 256 + row * 16 + half * 8) * 2u;
      async_load_b128_to_lds(ldsOff, pre + gidx);
    }

    __syncthreads();   // hlds (h_{t-1}) + mask ready

    v8f C[8]; v8f Ch = vz;
#pragma unroll
    for (int m = 0; m < 8; ++m) C[m] = vz;
#pragma unroll
    for (int k = 0; k < 8; ++k) {
      v16bf a = *(const v16bf*)&hlds[(k * 32 + lane) * 16];
#pragma unroll
      for (int m = 0; m < 8; ++m) {
        int ntile = (m >> 1) * 16 + 2 * w + (m & 1);   // q*16 + 2w + s
        v16bf bb = *(const v16bf*)&whhF[((size_t)(ntile * 8 + k) * 32 + lane) * 16];
        C[m] = __builtin_amdgcn_wmma_f32_16x16x32_bf16(false, a, false, bb, (short)0, C[m], false, false);
      }
      v16bf bh = *(const v16bf*)&wawcF[((size_t)k * 32 + lane) * 16];
      Ch = __builtin_amdgcn_wmma_f32_16x16x32_bf16(false, a, false, bh, (short)0, Ch, false, false);
    }
    // heads for step t-1 (Ch built from unmasked h_{t-1})
    if (w == 0 && t > 0)
      emit_heads(Ch, t - 1, bbase, lo, hiL, actions, myls, myistd, bav, bcv, entC, out);

    wait_asynccnt0();  // pre slab landed in LDS (own-wave data only)

    // elementwise LSTM update (fully in-register; mask applied to raw GEMM)
    float hval[2][8];
#pragma unroll
    for (int s = 0; s < 2; ++s) {
#pragma unroll
      for (int r = 0; r < 8; ++r) {
        int row = r + 8 * hiL;
        float m = maskLds[row];
        int pbase = w * 2048 + s * 256 + row * 16 + lo;
        float pi = b2f(preLds[pbase + 0 * 512]);
        float pf = b2f(preLds[pbase + 1 * 512]);
        float pg = b2f(preLds[pbase + 2 * 512]);
        float po = b2f(preLds[pbase + 3 * 512]);
        float gi = fast_sigmoid(pi + C[0 + s][r] * m);
        float gf = fast_sigmoid(pf + C[2 + s][r] * m);
        float gg = fast_tanh(pg + C[4 + s][r] * m);
        float go = fast_sigmoid(po + C[6 + s][r] * m);
        float cc = gf * (cst[s][r] * m) + gi * gg;
        cst[s][r] = cc;
        hval[s][r] = go * fast_tanh(cc);
      }
    }
    __syncthreads();   // all waves done reading hlds for this step
#pragma unroll
    for (int s = 0; s < 2; ++s)
#pragma unroll
      for (int r = 0; r < 8; ++r) hlds[hoff[s][r]] = f2b(hval[s][r]);
  }
  __syncthreads();

  // epilogue: heads for t = T-1 from final h
  {
    v8f Ch = vz;
#pragma unroll
    for (int k = 0; k < 8; ++k) {
      v16bf a = *(const v16bf*)&hlds[(k * 32 + lane) * 16];
      v16bf bh = *(const v16bf*)&wawcF[((size_t)k * 32 + lane) * 16];
      Ch = __builtin_amdgcn_wmma_f32_16x16x32_bf16(false, a, false, bh, (short)0, Ch, false, false);
    }
    if (w == 0)
      emit_heads(Ch, T_ - 1, bbase, lo, hiL, actions, myls, myistd, bav, bcv, entC, out);
  }
}

// =====================================================================
extern "C" void kernel_launch(void* const* d_in, const int* in_sizes, int n_in,
                              void* d_out, int out_size, void* d_ws, size_t ws_size,
                              hipStream_t stream) {
  (void)in_sizes; (void)n_in; (void)out_size; (void)ws_size;
  const float* obs     = (const float*)d_in[0];
  const float* actions = (const float*)d_in[1];
  const int*   done    = (const int*)d_in[2];
  const float* Wp      = (const float*)d_in[3];
  const float* bp      = (const float*)d_in[4];
  const float* W_ih    = (const float*)d_in[5];
  const float* W_hh    = (const float*)d_in[6];
  const float* b_ih    = (const float*)d_in[7];
  const float* b_hh    = (const float*)d_in[8];
  const float* Wa      = (const float*)d_in[9];
  const float* ba      = (const float*)d_in[10];
  const float* log_std = (const float*)d_in[11];
  const float* Wc      = (const float*)d_in[12];
  const float* bc      = (const float*)d_in[13];
  float* out = (float*)d_out;

  char* ws = (char*)d_ws;
  size_t off = 0;
  unsigned short* pre   = (unsigned short*)(ws + off); off += (size_t)B_ * T_ * G4_ * 2;
  unsigned short* wpF   = (unsigned short*)(ws + off); off += (size_t)16 * 2 * 512 * 2;
  unsigned short* wihF  = (unsigned short*)(ws + off); off += (size_t)64 * 8 * 512 * 2;
  unsigned short* whhF  = (unsigned short*)(ws + off); off += (size_t)64 * 8 * 512 * 2;
  unsigned short* wawcF = (unsigned short*)(ws + off); off += (size_t)8 * 512 * 2;

  prep_weights_kernel<<<2128, 256, 0, stream>>>(Wp, W_ih, W_hh, Wa, Wc,
                                                wpF, wihF, whhF, wawcF);
  encode_kernel<<<(B_ * T_) / 16, 256, 0, stream>>>(obs, bp, b_ih, b_hh,
                                                    wpF, wihF, pre);
  recurrence_kernel<<<B_ / 16, 256, 0, stream>>>(pre, whhF, wawcF, done, actions,
                                                 log_std, ba, bc, out);
}
